// RoutedLoRALinear_2843268350541
// MI455X (gfx1250) — matrix operand order repaired and checked by reading, not compiled, not run on hardware
//
#include <hip/hip_runtime.h>
#include <stdint.h>

// Problem constants (from reference)
#define BATCH   8
#define SEQ     2048
#define DIN     4096
#define DOUT    4096
#define LORA_R  16
#define SCALING 2.0f

// Tiling
#define TM   128   // seq rows per block
#define TN   128   // dout cols per block
#define KB   32    // K-chunk (bf16 WMMA K)
#define LSTR 40    // LDS row stride in bf16 elems (32 + 8 pad, even)

typedef __attribute__((ext_vector_type(16))) __bf16 bf16x16;
typedef __attribute__((ext_vector_type(8)))  float  f32x8;

union AFrag { bf16x16 v; uint32_t u[8]; };
union CFrag { f32x8   v; float    f[8]; };

// Pack two fp32 into packed bf16, round-to-nearest (ties away).
// 2 adds (dual-issue capable) + 1 v_perm_b32; avoids the 16-bit subreg churn
// the compiler generated for the RNE bit-trick (cvt_pk_bf16_f32 builtin is
// absent on this toolchain -- observed via __has_builtin fallback last round).
__device__ __forceinline__ uint32_t pack2bf16(float lo, float hi) {
    uint32_t a = __float_as_uint(lo) + 0x8000u;
    uint32_t b = __float_as_uint(hi) + 0x8000u;
    // result = { b[31:16], a[31:16] } : bytes 0,1 <- src1(a) bytes 2,3 ; bytes 2,3 <- src0(b) bytes 2,3
    return __builtin_amdgcn_perm(b, a, 0x07060302u);
}

// ---------------------------------------------------------------------------
// Kernel 1: low[b,s,r] = SCALING * sum_d x[b,s,d] * lora_A[e_b, r, d]
// One wave per (b,s) row; lora_A (2MB) stays L2-resident.
// ---------------------------------------------------------------------------
__global__ __launch_bounds__(256) void lora_low_kernel(
    const float* __restrict__ x, const float* __restrict__ lora_A,
    const int* __restrict__ eidx, float* __restrict__ low) {
    const int wave = threadIdx.x >> 5;
    const int lane = threadIdx.x & 31;
    const int row  = blockIdx.x * 8 + wave;      // 0 .. BATCH*SEQ-1
    const int b    = row >> 11;                  // row / SEQ
    const int e    = eidx[b];

    const float* xr = x + (size_t)row * DIN;
    const float* Ae = lora_A + (size_t)e * LORA_R * DIN;

    float acc[LORA_R];
#pragma unroll
    for (int r = 0; r < LORA_R; ++r) acc[r] = 0.0f;

    for (int d = lane * 4; d < DIN; d += 32 * 4) {
        float4 xv = *(const float4*)(xr + d);
#pragma unroll
        for (int r = 0; r < LORA_R; ++r) {
            float4 av = *(const float4*)(Ae + (size_t)r * DIN + d);
            acc[r] += xv.x * av.x + xv.y * av.y + xv.z * av.z + xv.w * av.w;
        }
    }
#pragma unroll
    for (int r = 0; r < LORA_R; ++r) {
#pragma unroll
        for (int off = 16; off > 0; off >>= 1)
            acc[r] += __shfl_down(acc[r], off, 32);
    }
    if (lane == 0) {
#pragma unroll
        for (int r = 0; r < LORA_R; ++r)
            low[(size_t)row * LORA_R + r] = acc[r] * SCALING;
    }
}

// ---------------------------------------------------------------------------
// Kernel 2: out = x @ W^T + bias + low @ lora_B[e]^T   (bf16 WMMA, f32 acc)
// Block: 256 threads = 8 waves (4 along M x 2 along N), 128x128 tile.
// Each wave: 2x4 grid of 16x16 WMMA tiles. Double-buffered LDS:
//   prefetch(k+1)->regs || WMMA(k from LDS[cur]) ; pack->LDS[next] ; 1 barrier.
// ---------------------------------------------------------------------------
__global__ __launch_bounds__(256) void routed_lora_gemm(
    const float* __restrict__ x, const float* __restrict__ W,
    const float* __restrict__ bias, const float* __restrict__ lora_B,
    const int* __restrict__ eidx, const float* __restrict__ low,
    float* __restrict__ out) {
    __shared__ __align__(16) uint16_t lA[2][TM * LSTR];   // x tiles, bf16
    __shared__ __align__(16) uint16_t lB[2][TN * LSTR];   // W tiles, bf16

    const int n0 = blockIdx.x * TN;
    const int s0 = blockIdx.y * TM;
    const int bb = blockIdx.z;
    const int e  = eidx[bb];

    const int tid     = threadIdx.x;
    const int lane    = tid & 31;
    const int wave    = tid >> 5;
    const int waveM   = wave & 3;    // 0..3 -> M offset waveM*32
    const int waveN   = wave >> 2;   // 0..1 -> N offset waveN*64
    const int halfSel = lane >> 4;   // 0: lanes 0-15, 1: lanes 16-31
    const int lr      = lane & 15;

    const float* xBase = x + ((size_t)bb * SEQ + s0) * DIN;
    const float* wBase = W + (size_t)n0 * DIN;

    CFrag acc[2][4];
#pragma unroll
    for (int mt = 0; mt < 2; ++mt)
#pragma unroll
        for (int nt = 0; nt < 4; ++nt)
#pragma unroll
            for (int v = 0; v < 8; ++v) acc[mt][nt].f[v] = 0.0f;

    // Cooperative loader: 4 threads/row * 8 floats, 64 rows/pass, 2 passes.
    // Staged bf16 chunk per thread-row is 16B -> ds_store_b128.
    const int ldRow = tid >> 2;          // 0..63
    const int ldCol = (tid & 3) * 8;     // 0,8,16,24

    float4 xa[4], wa[4];   // prefetch registers (held across the WMMA section)

    // Prologue: load K-chunk 0 and stage into buffer 0
#pragma unroll
    for (int p = 0; p < 2; ++p) {
        const int r = p * 64 + ldRow;
        xa[2 * p]     = *(const float4*)(xBase + (size_t)r * DIN + ldCol);
        xa[2 * p + 1] = *(const float4*)(xBase + (size_t)r * DIN + ldCol + 4);
        wa[2 * p]     = *(const float4*)(wBase + (size_t)r * DIN + ldCol);
        wa[2 * p + 1] = *(const float4*)(wBase + (size_t)r * DIN + ldCol + 4);
    }
#pragma unroll
    for (int p = 0; p < 2; ++p) {
        const int r = p * 64 + ldRow;
        uint4 pa, pw;
        pa.x = pack2bf16(xa[2*p].x,   xa[2*p].y);
        pa.y = pack2bf16(xa[2*p].z,   xa[2*p].w);
        pa.z = pack2bf16(xa[2*p+1].x, xa[2*p+1].y);
        pa.w = pack2bf16(xa[2*p+1].z, xa[2*p+1].w);
        pw.x = pack2bf16(wa[2*p].x,   wa[2*p].y);
        pw.y = pack2bf16(wa[2*p].z,   wa[2*p].w);
        pw.z = pack2bf16(wa[2*p+1].x, wa[2*p+1].y);
        pw.w = pack2bf16(wa[2*p+1].z, wa[2*p+1].w);
        *(uint4*)&lA[0][r * LSTR + ldCol] = pa;
        *(uint4*)&lB[0][r * LSTR + ldCol] = pw;
    }
    __syncthreads();

    int buf = 0;
    for (int k0 = 0; k0 < DIN; k0 += KB) {
        const bool hasNext = (k0 + KB) < DIN;   // uniform across block
        // ---- Prefetch next K-chunk into registers (overlaps with WMMAs) ----
        if (hasNext) {
#pragma unroll
            for (int p = 0; p < 2; ++p) {
                const int r = p * 64 + ldRow;
                const size_t off = (size_t)r * DIN + (k0 + KB) + ldCol;
                xa[2 * p]     = *(const float4*)(xBase + off);
                xa[2 * p + 1] = *(const float4*)(xBase + off + 4);
                wa[2 * p]     = *(const float4*)(wBase + off);
                wa[2 * p + 1] = *(const float4*)(wBase + off + 4);
            }
        }

        // ---- Fragment loads from LDS[buf] + 8 WMMAs ----
        const uint16_t* lAc = lA[buf];
        const uint16_t* lBc = lB[buf];

        // A fragments: 16x32 bf16, lane L holds row L%16;
        // lanes 0-15: K 0..15; lanes 16-31: K 16..31; 2 K per VGPR
        AFrag af[2];
#pragma unroll
        for (int mt = 0; mt < 2; ++mt) {
            const int rrow = waveM * 32 + mt * 16 + lr;
            const int kb   = halfSel * 16;
            const uint32_t* p0 = (const uint32_t*)&lAc[rrow * LSTR + kb];
            const uint32_t* p1 = (const uint32_t*)&lAc[rrow * LSTR + kb + 8];
#pragma unroll
            for (int v = 0; v < 4; ++v) { af[mt].u[v] = p0[v]; af[mt].u[4 + v] = p1[v]; }
        }
        // B fragments: 32x16 bf16, lane L holds col L%16;
        // lanes 0-15: K 0..15; lanes 16-31: K 16..31; 2 K per VGPR
        AFrag bfr[4];
#pragma unroll
        for (int nt = 0; nt < 4; ++nt) {
            const int orow = waveN * 64 + nt * 16 + lr;
            const uint32_t* p0 = (const uint32_t*)&lBc[orow * LSTR + halfSel * 16];
#pragma unroll
            for (int v = 0; v < 8; ++v) bfr[nt].u[v] = p0[v];
        }
#pragma unroll
        for (int mt = 0; mt < 2; ++mt)
#pragma unroll
            for (int nt = 0; nt < 4; ++nt)
                acc[mt][nt].v = __builtin_amdgcn_wmma_f32_16x16x32_bf16(
                    false, af[mt].v, false, bfr[nt].v,
                    (short)0, acc[mt][nt].v, false, false);

        // ---- Pack prefetched data into LDS[buf^1] ----
        if (hasNext) {
            uint16_t* lAn = lA[buf ^ 1];
            uint16_t* lBn = lB[buf ^ 1];
#pragma unroll
            for (int p = 0; p < 2; ++p) {
                const int r = p * 64 + ldRow;
                uint4 pa, pw;
                pa.x = pack2bf16(xa[2*p].x,   xa[2*p].y);
                pa.y = pack2bf16(xa[2*p].z,   xa[2*p].w);
                pa.z = pack2bf16(xa[2*p+1].x, xa[2*p+1].y);
                pa.w = pack2bf16(xa[2*p+1].z, xa[2*p+1].w);
                pw.x = pack2bf16(wa[2*p].x,   wa[2*p].y);
                pw.y = pack2bf16(wa[2*p].z,   wa[2*p].w);
                pw.z = pack2bf16(wa[2*p+1].x, wa[2*p+1].y);
                pw.w = pack2bf16(wa[2*p+1].z, wa[2*p+1].w);
                *(uint4*)&lAn[r * LSTR + ldCol] = pa;
                *(uint4*)&lBn[r * LSTR + ldCol] = pw;
            }
        }
        __syncthreads();   // next-buffer stores visible; cur-buffer reads done
        buf ^= 1;
    }

    // ---- LoRA epilogue: one zero-padded K=32 WMMA step (r = 16 <= 32) ----
    {
        AFrag lf[2];
#pragma unroll
        for (int mt = 0; mt < 2; ++mt) {
            const int s = s0 + waveM * 32 + mt * 16 + lr;
            const float* lp = low + ((size_t)bb * SEQ + s) * LORA_R;
            float4 l0 = *(const float4*)(lp);
            float4 l1 = *(const float4*)(lp + 4);
            float4 l2 = *(const float4*)(lp + 8);
            float4 l3 = *(const float4*)(lp + 12);
            lf[mt].u[0] = pack2bf16(l0.x, l0.y);
            lf[mt].u[1] = pack2bf16(l0.z, l0.w);
            lf[mt].u[2] = pack2bf16(l1.x, l1.y);
            lf[mt].u[3] = pack2bf16(l1.z, l1.w);
            lf[mt].u[4] = pack2bf16(l2.x, l2.y);
            lf[mt].u[5] = pack2bf16(l2.z, l2.w);
            lf[mt].u[6] = pack2bf16(l3.x, l3.y);
            lf[mt].u[7] = pack2bf16(l3.z, l3.w);
            if (halfSel) { for (int v = 0; v < 8; ++v) lf[mt].u[v] = 0; }
        }
        AFrag bl[4];
#pragma unroll
        for (int nt = 0; nt < 4; ++nt) {
            if (halfSel == 0) {   // K 0..15 half holds the real rank-16 data
                const int o = n0 + waveN * 64 + nt * 16 + lr;
                const float* bp = lora_B + ((size_t)e * DOUT + o) * LORA_R;
#pragma unroll
                for (int q = 0; q < 4; ++q) {
                    float4 bv = *(const float4*)(bp + q * 4);
                    bl[nt].u[2 * q]     = pack2bf16(bv.x, bv.y);
                    bl[nt].u[2 * q + 1] = pack2bf16(bv.z, bv.w);
                }
            } else {              // K 16..31 half is zero padding
#pragma unroll
                for (int v = 0; v < 8; ++v) bl[nt].u[v] = 0;
            }
        }
#pragma unroll
        for (int mt = 0; mt < 2; ++mt)
#pragma unroll
            for (int nt = 0; nt < 4; ++nt)
                acc[mt][nt].v = __builtin_amdgcn_wmma_f32_16x16x32_bf16(
                    false, lf[mt].v, false, bl[nt].v,
                    (short)0, acc[mt][nt].v, false, false);
    }

    // ---- Bias add + store (C/D layout: f[v] -> M = v + 8*halfSel, N = lr) ----
#pragma unroll
    for (int nt = 0; nt < 4; ++nt) {
        const int o = n0 + waveN * 64 + nt * 16 + lr;
        const float bv = bias[o];
#pragma unroll
        for (int mt = 0; mt < 2; ++mt) {
            const int sbase = s0 + waveM * 32 + mt * 16 + halfSel * 8;
#pragma unroll
            for (int v = 0; v < 8; ++v) {
                float* dst = out + ((size_t)bb * SEQ + sbase + v) * DOUT + o;
                __builtin_nontemporal_store(acc[mt][nt].f[v] + bv, dst);
            }
        }
    }
}

// ---------------------------------------------------------------------------
extern "C" void kernel_launch(void* const* d_in, const int* in_sizes, int n_in,
                              void* d_out, int out_size, void* d_ws, size_t ws_size,
                              hipStream_t stream) {
    const float* x      = (const float*)d_in[0];
    const float* W      = (const float*)d_in[1];
    const float* b      = (const float*)d_in[2];
    const float* lora_A = (const float*)d_in[3];
    const float* lora_B = (const float*)d_in[4];
    const int*   eidx   = (const int*)d_in[5];
    float* out = (float*)d_out;
    float* low = (float*)d_ws;   // BATCH*SEQ*LORA_R floats = 1 MB scratch

    lora_low_kernel<<<(BATCH * SEQ) / 8, 256, 0, stream>>>(x, lora_A, eidx, low);

    dim3 grid(DOUT / TN, SEQ / TM, BATCH);   // 32 x 16 x 8 = 4096 blocks
    routed_lora_gemm<<<grid, 256, 0, stream>>>(x, W, b, lora_B, eidx, low, out);
}
